// _WaveRNNInferrer_36696200577054
// MI455X (gfx1250) — matrix operand order repaired and checked by reading, not compiled, not run on hardware
//
#include <hip/hip_runtime.h>
#include <hip/hip_bf16.h>

// ---------------------------------------------------------------------------
// WaveRNN inference, persistent-kernel design for MI455X (gfx1250).
//   HIDDEN=896, HALF=448, BINS=256, T=16384
//   blocks 0..20 : hp = Whh @ h   (168 waves, one 16-row tile each, weights
//                  LDS-resident as pre-swizzled bf16 WMMA A-fragments)
//   block 21     : coarse tail (gates -> pre(448x448) -> bins(256x448) -> argmax)
//                  bin-fragments LDS-resident; pre-fragments streamed from L2
//                  with register-preloaded fragment clauses.
//   block 22     : fine tail (same structure)
//   3 grid syncs per step via monotonic atomic counter.
// ---------------------------------------------------------------------------

typedef __attribute__((ext_vector_type(16))) __bf16 v16bf;
typedef __attribute__((ext_vector_type(8)))  float  v8f;

#define T_STEPS 16384
#define HIDDEN  896
#define HALF    448
#define BINS    256
#define NMAT    21
#define NBLK    23
#define KC_HID  28          // 896/32
#define KC_HALF 14          // 448/32

#define LDSW_BYTES     (8 * KC_HID * 512 * 2)        // 229376: 8 waves x 28 kc x 512 bf16
#define MAT_SMEM       (LDSW_BYTES + HIDDEN * 2)     // + bf16 h

#define TAIL_BIN_BYTES (16 * KC_HALF * 512 * 2)      // 229376: bin frags, LDS-resident
#define TAIL_HNEW      (TAIL_BIN_BYTES)              // ushort[448]
#define TAIL_PRE       (TAIL_BIN_BYTES + 1024)       // ushort[448]
#define TAIL_REDV      (TAIL_BIN_BYTES + 2048)       // float[256]
#define TAIL_REDI      (TAIL_BIN_BYTES + 3072)       // int[256]
#define SMEM_BYTES     (TAIL_BIN_BYTES + 4096)       // 233472 (>= MAT_SMEM)

__device__ __forceinline__ unsigned short bf16bits(float x) {
    union { float f; unsigned u; } v; v.f = x;
    unsigned r = v.u + 0x7FFFu + ((v.u >> 16) & 1u);   // round-to-nearest-even
    return (unsigned short)(r >> 16);
}

__device__ __forceinline__ v8f wmma_bf16(v16bf a, v16bf b, v8f c) {
    // D = A(16x32 bf16) x B(32x16 bf16) + C(16x16 f32)
    return __builtin_amdgcn_wmma_f32_16x16x32_bf16(
        /*neg_a*/false, a, /*neg_b*/false, b,
        /*c_mod*/(short)0, c, /*reuse_a*/false, /*reuse_b*/false);
}

__device__ __forceinline__ void grid_sync(int* counter, unsigned target) {
    __threadfence();              // release our global writes (agent scope)
    __syncthreads();
    if (threadIdx.x == 0) {
        __hip_atomic_fetch_add(counter, 1, __ATOMIC_RELEASE, __HIP_MEMORY_SCOPE_AGENT);
        while ((unsigned)__hip_atomic_load(counter, __ATOMIC_ACQUIRE,
                                           __HIP_MEMORY_SCOPE_AGENT) < target)
            __builtin_amdgcn_s_sleep(2);
    }
    __syncthreads();
    __threadfence();              // acquire: invalidate stale L0 before reads
}

// ---- one autoregressive half (coarse: fine==false, fine: fine==true) ------
__device__ void tail_half(int t, bool fine,
                          const float* __restrict__ Win,      // [1344 x (2|3)]
                          const float* __restrict__ biasRow,  // [1344] (row t)
                          const float* __restrict__ hp,       // [1344] slice
                          float*       __restrict__ hslice,   // [448] slice of h
                          const unsigned short* __restrict__ preFrag, // global
                          const float* __restrict__ preBias,
                          const float* __restrict__ binBias,
                          float* cf, float* outSlot, char* smem)
{
    const int tid  = threadIdx.x;
    const int lane = tid & 31;
    const int wave = tid >> 5;
    const unsigned short* binLds = (const unsigned short*)smem;   // LDS-resident
    unsigned short* hnew_bf = (unsigned short*)(smem + TAIL_HNEW);
    unsigned short* pre_bf  = (unsigned short*)(smem + TAIL_PRE);
    float* redv = (float*)(smem + TAIL_REDV);
    int*   redi = (int*)  (smem + TAIL_REDI);

    const float c_prev = cf[0], f_prev = cf[1], c_cur = cf[2];

    // ---- GRU gates (per-element independent) ----
    for (int j = tid; j < HALF; j += 256) {
        float pr, pu, pn;
        if (!fine) {
            pr = biasRow[j]        + Win[(j)*2+0]*c_prev        + Win[(j)*2+1]*f_prev;
            pu = biasRow[HALF+j]   + Win[(HALF+j)*2+0]*c_prev   + Win[(HALF+j)*2+1]*f_prev;
            pn = biasRow[2*HALF+j] + Win[(2*HALF+j)*2+0]*c_prev + Win[(2*HALF+j)*2+1]*f_prev;
        } else {
            pr = biasRow[j]        + Win[(j)*3+0]*c_prev        + Win[(j)*3+1]*f_prev        + Win[(j)*3+2]*c_cur;
            pu = biasRow[HALF+j]   + Win[(HALF+j)*3+0]*c_prev   + Win[(HALF+j)*3+1]*f_prev   + Win[(HALF+j)*3+2]*c_cur;
            pn = biasRow[2*HALF+j] + Win[(2*HALF+j)*3+0]*c_prev + Win[(2*HALF+j)*3+1]*f_prev + Win[(2*HALF+j)*3+2]*c_cur;
        }
        float r = 1.f / (1.f + expf(-(pr + hp[j])));
        float u = 1.f / (1.f + expf(-(pu + hp[HALF + j])));
        float n = tanhf(pn + r * hp[2*HALF + j]);
        float hn = u * (hslice[j] - n) + n;
        hslice[j]  = hn;                 // new hidden half -> global
        hnew_bf[j] = bf16bits(hn);       // bf16 copy for WMMA B operand
    }
    __syncthreads();

    // ---- pre = relu(pre_b + pre_W(448x448) @ h_new) ----
    // Preload all 14 A-fragments of a tile into registers (one load clause),
    // then drain with WMMAs so L2 latency is paid once per tile, not per kc.
    for (int tile = wave; tile < 28; tile += 8) {
        const unsigned short* base = preFrag + (size_t)(tile*KC_HALF)*512 + lane*16;
        v16bf a[KC_HALF];
        #pragma unroll
        for (int kc = 0; kc < KC_HALF; ++kc)
            a[kc] = *(const v16bf*)(base + kc*512);
        v8f acc = {};
        #pragma unroll
        for (int kc = 0; kc < KC_HALF; ++kc) {
            v16bf b = *(const v16bf*)(hnew_bf + kc*32 + ((lane >= 16) ? 16 : 0));
            acc = wmma_bf16(a[kc], b, acc);
        }
        if ((lane & 15) == 0) {
            int rbase = tile*16 + ((lane >= 16) ? 8 : 0);
            #pragma unroll
            for (int r = 0; r < 8; ++r) {
                float p = preBias[rbase + r] + acc[r];
                pre_bf[rbase + r] = bf16bits(p > 0.f ? p : 0.f);
            }
        }
    }
    __syncthreads();

    // ---- logits = bin_b + bin_W(256x448) @ pre  (LDS-resident fragments) ----
    for (int tile = wave*2; tile < wave*2 + 2; ++tile) {
        const unsigned short* base = binLds + (tile*KC_HALF)*512 + lane*16;
        v8f acc = {};
        #pragma unroll
        for (int kc = 0; kc < KC_HALF; ++kc) {
            v16bf a = *(const v16bf*)(base + kc*512);
            v16bf b = *(const v16bf*)(pre_bf + kc*32 + ((lane >= 16) ? 16 : 0));
            acc = wmma_bf16(a, b, acc);
        }
        if ((lane & 15) == 0) {
            int rbase = tile*16 + ((lane >= 16) ? 8 : 0);
            #pragma unroll
            for (int r = 0; r < 8; ++r) redv[rbase + r] = binBias[rbase + r] + acc[r];
        }
    }
    __syncthreads();

    // ---- argmax over 256 logits (first-max-index ties) ----
    redi[tid] = tid;
    __syncthreads();
    for (int s = 128; s > 0; s >>= 1) {
        if (tid < s) {
            float a = redv[tid], b = redv[tid + s];
            int   ia = redi[tid], ib = redi[tid + s];
            if (b > a || (b == a && ib < ia)) { redv[tid] = b; redi[tid] = ib; }
        }
        __syncthreads();
    }
    if (tid == 0) {
        int samp = redi[0];
        *outSlot = (float)samp;
        float scaled = (float)samp * (2.0f / 255.0f) - 1.0f;
        if (!fine) cf[2] = scaled;                  // feeds fine input this step
        else { cf[0] = cf[2]; cf[1] = scaled; }     // becomes next step's prev
    }
    // ---- prefetch next timestep's bias row (cold HBM read otherwise):
    // 1344 floats = 5376 B = 42 x 128B lines; hidden behind 2 syncs + phase A.
    if (t + 1 < T_STEPS && tid < 42)
        __builtin_prefetch(biasRow + 1344 + tid*32, 0, 1);
}

// ---------------------------------------------------------------------------
__launch_bounds__(256, 1)
__global__ void wavernn_persist(
    const float* __restrict__ Whh,  const float* __restrict__ bh,
    const float* __restrict__ Wc,   const float* __restrict__ Wf,
    const float* __restrict__ cbias,const float* __restrict__ fbias,
    const float* __restrict__ prebC,const float* __restrict__ binbC,
    const float* __restrict__ prebF,const float* __restrict__ binbF,
    const unsigned short* __restrict__ preFragC, const unsigned short* __restrict__ preFragF,
    const unsigned short* __restrict__ binFragC, const unsigned short* __restrict__ binFragF,
    float* h, float* hp, float* cf, int* counter, float* out)
{
    extern __shared__ char smem[];
    const int blk = blockIdx.x, tid = threadIdx.x;
    const int wave = tid >> 5, lane = tid & 31;
    const bool isMat = (blk < NMAT);

    unsigned short* ldsW = (unsigned short*)smem;                // weight frags
    unsigned short* ldsH = (unsigned short*)(smem + LDSW_BYTES); // bf16 h [896]

    // ---- one-time staging ----
    if (isMat) {
        // Whh tiles -> LDS as WMMA A-fragments.
        // 16-bit A 16x32 layout: lane l holds row M=l%16; its 16
        // elements are K = kc*32 + (l>=16?16:0) + {0..15} (contiguous, matches B).
        const int tile = blk*8 + wave;                 // 0..167
        for (int kc = 0; kc < KC_HID; ++kc) {
            const int row = tile*16 + (lane & 15);
            const int kb  = kc*32 + ((lane >= 16) ? 16 : 0);
            const float* src = Whh + (size_t)row*HIDDEN + kb;
            unsigned short* dst = ldsW + wave*(KC_HID*512) + kc*512 + lane*16;
            #pragma unroll
            for (int e = 0; e < 8; ++e) {
                dst[e]     = bf16bits(src[e]);
                dst[8 + e] = bf16bits(src[8 + e]);
            }
        }
    } else {
        // bin fragments -> LDS (read every step; 229KB stays resident)
        const unsigned short* src = (blk == NMAT) ? binFragC : binFragF;
        uint4* dst = (uint4*)smem;
        const uint4* s4 = (const uint4*)src;
        for (int i = tid; i < TAIL_BIN_BYTES/16; i += 256) dst[i] = s4[i];
    }
    __syncthreads();

    // hoist hp-bias loads out of the 16384-step loop (lanes 0/16 only use them)
    float bias8[8] = {0,0,0,0,0,0,0,0};
    if (isMat && (lane & 15) == 0) {
        const int rbase = (blk*8 + wave)*16 + ((lane >= 16) ? 8 : 0);
        #pragma unroll
        for (int r = 0; r < 8; ++r) bias8[r] = bh[rbase + r];
    }

    unsigned gen = 0;
    #pragma unroll 1
    for (int t = 0; t < T_STEPS; ++t) {
        // ---- phase A: hp = bh + Whh @ h  (168 waves, WMMA, LDS weights) ----
        if (isMat) {
            for (int j = tid; j < HIDDEN; j += 256) ldsH[j] = bf16bits(h[j]);
            __syncthreads();
            const int tile = blk*8 + wave;
            const unsigned short* wb = ldsW + wave*(KC_HID*512) + lane*16;
            const unsigned short* hb = ldsH + ((lane >= 16) ? 16 : 0);
            // 2-deep pipeline: next A-frag ds_load issues before current WMMA
            v16bf a0 = *(const v16bf*)(wb);
            v8f acc = {};
            #pragma unroll
            for (int kc = 0; kc < KC_HID; ++kc) {
                v16bf a1 = a0;
                if (kc + 1 < KC_HID) a1 = *(const v16bf*)(wb + (kc+1)*512);
                v16bf b = *(const v16bf*)(hb + kc*32);
                acc = wmma_bf16(a0, b, acc);
                a0 = a1;
            }
            if ((lane & 15) == 0) {     // column 0 lanes: rows tile*16+{0..7|8..15}
                const int rbase = tile*16 + ((lane >= 16) ? 8 : 0);
                #pragma unroll
                for (int r = 0; r < 8; ++r) hp[rbase + r] = bias8[r] + acc[r];
            }
        }
        ++gen; grid_sync(counter, (unsigned)NBLK * gen);

        // ---- phase B: coarse half ----
        if (blk == NMAT)
            tail_half(t, false, Wc, cbias + (size_t)t*1344, hp, h,
                      preFragC, prebC, binbC, cf, out + t, smem);
        ++gen; grid_sync(counter, (unsigned)NBLK * gen);

        // ---- phase C: fine half (needs coarse sample) ----
        if (blk == NMAT + 1)
            tail_half(t, true, Wf, fbias + (size_t)t*1344, hp + 1344, h + HALF,
                      preFragF, prebF, binbF, cf, out + T_STEPS + t, smem);
        ++gen; grid_sync(counter, (unsigned)NBLK * gen);
    }

    if (blk == 0)
        for (int j = tid; j < HIDDEN; j += 256) out[2*T_STEPS + j] = h[j];
}

// ---- setup: swizzle fp32 weights into bf16 WMMA A-fragment order ----------
__global__ void swizzle_frags(const float* __restrict__ W, unsigned short* __restrict__ out,
                              int tiles, int kchunks, int K)
{
    const int total = tiles * kchunks * 512;
    for (int idx = blockIdx.x*blockDim.x + threadIdx.x; idx < total;
         idx += gridDim.x*blockDim.x) {
        int e  = idx & 15;
        int l  = (idx >> 4) & 31;
        int kc = (idx >> 9) % kchunks;
        int tl = idx / (kchunks * 512);
        int row = tl*16 + (l & 15);
        int k   = kc*32 + ((l >= 16) ? 16 : 0) + e;
        out[idx] = bf16bits(W[row*K + k]);
    }
}

__global__ void init_state(int* counter, float* h, float* cf, const float* h0) {
    int i = blockIdx.x*blockDim.x + threadIdx.x;
    if (i < HIDDEN) h[i] = h0[i];
    if (i == 0) {
        counter[0] = 0;
        float iv = 128.0f * (2.0f / 255.0f) - 1.0f;   // (BINS/2)*INV_BINS - 1
        cf[0] = iv; cf[1] = iv; cf[2] = iv;
    }
}

// ---------------------------------------------------------------------------
extern "C" void kernel_launch(void* const* d_in, const int* in_sizes, int n_in,
                              void* d_out, int out_size, void* d_ws, size_t ws_size,
                              hipStream_t stream) {
    const float* Whh   = (const float*)d_in[0];
    const float* bh    = (const float*)d_in[1];
    const float* Wc    = (const float*)d_in[2];
    const float* Wf    = (const float*)d_in[3];
    const float* cbias = (const float*)d_in[4];
    const float* fbias = (const float*)d_in[5];
    const float* preWc = (const float*)d_in[6];
    const float* prebC = (const float*)d_in[7];
    const float* binWc = (const float*)d_in[8];
    const float* binbC = (const float*)d_in[9];
    const float* preWf = (const float*)d_in[10];
    const float* prebF = (const float*)d_in[11];
    const float* binWf = (const float*)d_in[12];
    const float* binbF = (const float*)d_in[13];
    const float* h0    = (const float*)d_in[14];
    float* out = (float*)d_out;

    // workspace layout (256B-aligned regions)
    char* ws = (char*)d_ws;
    int*   counter = (int*)ws;                         // 256 B
    float* h  = (float*)(ws + 256);                    // 896 f32
    float* hp = (float*)(ws + 3840);                   // 2688 f32
    float* cf = (float*)(ws + 14592);                  // 3 f32
    unsigned short* preFragC = (unsigned short*)(ws + 15360);    // 28*14*512
    unsigned short* preFragF = (unsigned short*)(ws + 416768);
    unsigned short* binFragC = (unsigned short*)(ws + 818176);   // 16*14*512
    unsigned short* binFragF = (unsigned short*)(ws + 1047552);

    (void)in_sizes; (void)n_in; (void)out_size; (void)ws_size;

    // allow >64KB dynamic LDS (320KB/WGP on CDNA5); idempotent, capture-safe
    (void)hipFuncSetAttribute((const void*)wavernn_persist,
                              hipFuncAttributeMaxDynamicSharedMemorySize, SMEM_BYTES);

    init_state<<<4, 256, 0, stream>>>(counter, h, cf, h0);
    swizzle_frags<<<256, 256, 0, stream>>>(preWc, preFragC, 28, 14, HALF);
    swizzle_frags<<<256, 256, 0, stream>>>(preWf, preFragF, 28, 14, HALF);
    swizzle_frags<<<128, 256, 0, stream>>>(binWc, binFragC, 16, 14, HALF);
    swizzle_frags<<<128, 256, 0, stream>>>(binWf, binFragF, 16, 14, HALF);

    wavernn_persist<<<NBLK, 256, SMEM_BYTES, stream>>>(
        Whh, bh, Wc, Wf, cbias, fbias, prebC, binbC, prebF, binbF,
        preFragC, preFragF, binFragC, binFragF, h, hp, cf, counter, out);
}
